// SoftHist_71579924955164
// MI455X (gfx1250) — matrix operand order repaired and checked by reading, not compiled, not run on hardware
//
#include <hip/hip_runtime.h>

// CDNA5 / gfx1250 soft-histogram.
// batch_hist[pixel,bin] = sum_b soft(x[b,pixel])[bin] is computed as a true
// matmul: soft[bin] = s_bin - s_{bin+1} (edge sigmoids telescope), so
//   batch_hist = A(16 pixels x K=32 edge-sigmoids) x M(constant 32x16, +/-1)
// via v_wmma_f32_16x16x32_f16, accumulated in f32 over 16 WMMAs (32 batches).

typedef __attribute__((ext_vector_type(16))) _Float16 v16h;
typedef __attribute__((ext_vector_type(8)))  float    v8f;

namespace {
constexpr int   kBins  = 16;
constexpr float kMinV  = -0.2f;
constexpr float kMaxV  = 10.0f;
constexpr float kSigma = 100.0f;
constexpr float kMom   = 0.1f;
constexpr float kDelta = (kMaxV - kMinV) / (float)kBins;   // 0.6375
constexpr int   kB = 32, kC = 32, kH = 128, kW = 128;
constexpr int   kCHW = kC * kH * kW;                        // 524288 pixels
constexpr float kLog2e = 1.4426950408889634f;
}

// s_i = sigmoid(sigma * (x - edge_i)); one HW transcendental when tanh exists.
#if __has_builtin(__builtin_amdgcn_tanhf)
#define EDGE_SCALE (0.5f * kSigma)                 // arg = t - E, s = .5 + .5*tanh(arg)
__device__ __forceinline__ float edge_sigmoid(float t, float E) {
  return __builtin_fmaf(0.5f, __builtin_amdgcn_tanhf(t - E), 0.5f);
}
#else
#define EDGE_SCALE (kSigma * kLog2e)               // s = rcp(1 + exp2(E - t))
__device__ __forceinline__ float edge_sigmoid(float t, float E) {
  return __builtin_amdgcn_rcpf(1.0f + __builtin_amdgcn_exp2f(E - t));
}
#endif

__global__ __launch_bounds__(256, 1)
void SoftHist_71579924955164_kernel(const float* __restrict__ in,
                                    const float* __restrict__ run,
                                    float* __restrict__ out) {
  const int lane  = (int)(threadIdx.x & 31u);
  const int gwave = (int)blockIdx.x * 8 + (int)(threadIdx.x >> 5);
  const int pixBase = gwave * 16;      // 16 pixels (M rows / A rows) per wave
  const int n    = lane & 15;          // pixel row for A; bin column for C/D
  const int half = lane >> 4;          // K-half select per 16-bit A layout

  // Constant B matrix (32x16 f16), same K order as A: lanes 0-15 hold
  // K in {0..7, 16..23}, lanes 16-31 hold K in {8..15, 24..31}. With
  // k = batch'*16 + edge, element e has edge == half*8 + e%8, so
  // B[e] = M[edge][n] = (edge==n) - (edge==n+1). soft[15] needs -s_16, but
  // s_16 = sigmoid(100*(x-10)) == 0 exactly in fp32 for all sane inputs.
  v16h bm;
#pragma unroll
  for (int e = 0; e < 16; ++e) {
    const int ed = half * 8 + (e & 7);
    float v = (ed == n ? 1.0f : 0.0f) - (ed == n + 1 ? 1.0f : 0.0f);
    bm[e] = (_Float16)v;
  }

  // Per-lane edge constants. 16-bit A 16x32 layout: lanes 0-15 hold
  // K in {0..7, 16..23}, lanes 16-31 hold K in {8..15, 24..31}; with
  // k = b'*16 + i, element e maps to batch b' = e/8, edge i = half*8 + e%8.
  float E[8];
#pragma unroll
  for (int j = 0; j < 8; ++j)
    E[j] = EDGE_SCALE * (kMinV + (float)(half * 8 + j) * kDelta);

  const float* __restrict__ xp = in + pixBase + n;   // pixel = n for A rows

  v8f acc = {};
#pragma unroll 4
  for (int it = 0; it < 16; ++it) {                  // 2 batches per WMMA
    const float x0 = xp[(size_t)(2 * it + 0) * kCHW];
    const float x1 = xp[(size_t)(2 * it + 1) * kCHW];
    const float t0 = EDGE_SCALE * x0;
    const float t1 = EDGE_SCALE * x1;
    v16h a;
#pragma unroll
    for (int j = 0; j < 8; ++j) {
      a[j]     = (_Float16)edge_sigmoid(t0, E[j]);   // batch b0, edges half*8+j
      a[8 + j] = (_Float16)edge_sigmoid(t1, E[j]);   // batch b1
    }
    // D[pixel, bin] += sum_k A[pixel,k] * M[k%16, bin]
    acc = __builtin_amdgcn_wmma_f32_16x16x32_f16(
        /*neg_a=*/false, a, /*neg_b=*/false, bm,
        /*c_mod=*/(short)0, acc, /*reuse_a=*/false, /*reuse_b=*/false);
  }

  // Epilogue. C/D layout: VGPR v -> row M=v (lanes 0-15) or M=v+8 (16-31),
  // column N = lane%16 = bin. Rows are pixels -> contiguous 64B per row-half.
  const float* __restrict__ rp = run + (size_t)pixBase * kBins;
  float* __restrict__ op = out + (size_t)pixBase * kBins;
#pragma unroll
  for (int v = 0; v < 8; ++v) {
    const int off = (v + half * 8) * kBins + n;
    // cur = (1-m)*running + m*batch_hist + 1
    float cur = __builtin_fmaf(1.0f - kMom, rp[off],
                __builtin_fmaf(kMom, acc[v], 1.0f));
    // normalize over bins == over the 16-lane group
    float sum = cur;
    sum += __shfl_xor(sum, 1, 16);
    sum += __shfl_xor(sum, 2, 16);
    sum += __shfl_xor(sum, 4, 16);
    sum += __shfl_xor(sum, 8, 16);
    op[off] = cur / sum;
  }
}

extern "C" void kernel_launch(void* const* d_in, const int* in_sizes, int n_in,
                              void* d_out, int out_size, void* d_ws, size_t ws_size,
                              hipStream_t stream) {
  (void)in_sizes; (void)n_in; (void)out_size; (void)d_ws; (void)ws_size;
  const float* in  = (const float*)d_in[0];   // [B,C,H,W] f32
  const float* run = (const float*)d_in[1];   // [C,H,W,16] f32
  float* out = (float*)d_out;                 // [C,H,W,16] f32
  // 16 pixels per wave, 8 waves per block -> 128 pixels/block, 4096 blocks.
  dim3 grid(kCHW / 128), block(256);
  SoftHist_71579924955164_kernel<<<grid, block, 0, stream>>>(in, run, out);
}
